// HGPSLModel_1348619731617
// MI455X (gfx1250) — matrix operand order repaired and checked by reading, not compiled, not run on hardware
//
#include <hip/hip_runtime.h>
#include <hip/hip_bf16.h>

// ---------------------------------------------------------------------------
// HGP-SL GCN forward for MI455X (gfx1250, wave32).
// All dense GEMMs run through v_wmma_f32_16x16x32_f16 (f32 accumulate).
// Block tile 128x64, 8 waves, 32x32 output per wave (4 WMMAs / K-step).
// A fragments = two contiguous ds_load_b128; B tile stored transposed in LDS
// so B fragments are also two contiguous ds_load_b128.
// ---------------------------------------------------------------------------

typedef _Float16 v16h __attribute__((ext_vector_type(16)));
typedef _Float16 v8h  __attribute__((ext_vector_type(8)));
typedef _Float16 v2h  __attribute__((ext_vector_type(2)));
typedef float    v8f  __attribute__((ext_vector_type(8)));
typedef float    v4f  __attribute__((ext_vector_type(4)));

#define AS_LD 40   // f16 row stride for A tile: 80 B, 16-byte aligned rows
#define BT_LD 40   // f16 row stride for transposed B tile

// ---------------------------------------------------------------------------
// Batched GEMM: C[M,N] = A[M,K] * B[K,N]  (f32 in/out, f16 WMMA, f32 acc)
// REQUIRES: K % 32 == 0, M % 128 == 0, N % 64 == 0, B rows 16B-aligned.
// Optional bias + relu epilogue. grid = (N/64, M/128, batch).
// ---------------------------------------------------------------------------
__global__ __launch_bounds__(256)
void k_wmma_gemm(const float* __restrict__ A, const float* __restrict__ Bm,
                 float* __restrict__ C, int M, int N, int K,
                 long lda, long ldb, long ldc,
                 long sA, long sB, long sC,
                 const float* __restrict__ bias, int relu)
{
    (void)M; (void)N;
    __shared__ alignas(16) _Float16 As[128 * AS_LD];   // 128x32 A tile
    __shared__ alignas(16) _Float16 BsT[64 * BT_LD];   // 32x64 B tile, transposed

    const int bz = blockIdx.z;
    A  += (long)bz * sA;
    Bm += (long)bz * sB;
    C  += (long)bz * sC;

    const int n0    = blockIdx.x * 64;
    const int m0    = blockIdx.y * 128;
    const int tid   = threadIdx.x;
    const int lane  = tid & 31;
    const int wave  = tid >> 5;
    const int mband = (wave & 3) * 32;    // 32-row band within 128
    const int nband = (wave >> 2) * 32;   // 32-col band within 64
    const int mrow  = lane & 15;
    const int half  = lane >> 4;

    v8f acc00 = {}, acc01 = {}, acc10 = {}, acc11 = {};

    for (int k0 = 0; k0 < K; k0 += 32) {
        // ---- Phase 1: global -> registers (no divergence, batched loads) ----
        // A tile: 128x32 floats as 2048 adjacent pairs; 8 pairs per thread.
        float a0v[8], a1v[8];
#pragma unroll
        for (int i = 0; i < 8; ++i) {
            int p = tid + i * 256;
            int r = p >> 4, c2 = (p & 15) * 2;
            const float* src = A + (long)(m0 + r) * lda + (k0 + c2);
            a0v[i] = src[0];
            a1v[i] = src[1];
        }
        // B tile: 32x64 floats as 512 float4 chunks; 2 per thread (16B aligned).
        v4f bv[2];
#pragma unroll
        for (int i = 0; i < 2; ++i) {
            int p = tid + i * 256;
            int r = p >> 4, c4 = (p & 15) * 4;
            bv[i] = *(const v4f*)(Bm + (long)(k0 + r) * ldb + (n0 + c4));
        }

        __syncthreads();   // previous iteration's fragment reads complete

        // ---- Phase 2: registers -> LDS (f32 -> f16) ----
#pragma unroll
        for (int i = 0; i < 8; ++i) {
            int p = tid + i * 256;
            int r = p >> 4, c2 = (p & 15) * 2;
            v2h pk;
            pk[0] = (_Float16)a0v[i];
            pk[1] = (_Float16)a1v[i];
            *(v2h*)&As[r * AS_LD + c2] = pk;
        }
#pragma unroll
        for (int i = 0; i < 2; ++i) {
            int p = tid + i * 256;
            int r = p >> 4, c4 = (p & 15) * 4;
#pragma unroll
            for (int j = 0; j < 4; ++j)
                BsT[(c4 + j) * BT_LD + r] = (_Float16)bv[i][j];
        }

        __syncthreads();   // tile visible to all waves

        // ---- Phase 3: fragments + WMMA ----
        // A frag (16x32 f16, ISA 7.12.2): lane (mrow,half) needs
        // k in [8*half, 8*half+7] and [16+8*half, ...+7]  -> two b128 chunks.
        const _Float16* Ap0 = &As[(mband + mrow) * AS_LD + 8 * half];
        const _Float16* Ap1 = &As[(mband + 16 + mrow) * AS_LD + 8 * half];
        v8h a0lo = *(const v8h*)Ap0;
        v8h a0hi = *(const v8h*)(Ap0 + 16);
        v8h a1lo = *(const v8h*)Ap1;
        v8h a1hi = *(const v8h*)(Ap1 + 16);
        v16h a0 = __builtin_shufflevector(a0lo, a0hi, 0,1,2,3,4,5,6,7,
                                          8,9,10,11,12,13,14,15);
        v16h a1 = __builtin_shufflevector(a1lo, a1hi, 0,1,2,3,4,5,6,7,
                                          8,9,10,11,12,13,14,15);
        // B frag (32x16 f16): lane (n=mrow,half) needs k in [8*half, 8*half+7]
        // and [16+8*half, ...+7] (same as A) -> two b128 chunks.
        const _Float16* Bp0 = &BsT[(nband + mrow) * BT_LD + 8 * half];
        const _Float16* Bp1 = &BsT[(nband + 16 + mrow) * BT_LD + 8 * half];
        v8h b0lo = *(const v8h*)Bp0;
        v8h b0hi = *(const v8h*)(Bp0 + 16);
        v8h b1lo = *(const v8h*)Bp1;
        v8h b1hi = *(const v8h*)(Bp1 + 16);
        v16h b0 = __builtin_shufflevector(b0lo, b0hi, 0,1,2,3,4,5,6,7,
                                          8,9,10,11,12,13,14,15);
        v16h b1 = __builtin_shufflevector(b1lo, b1hi, 0,1,2,3,4,5,6,7,
                                          8,9,10,11,12,13,14,15);

        acc00 = __builtin_amdgcn_wmma_f32_16x16x32_f16(false, a0, false, b0,
                                                       (short)0, acc00, false, false);
        acc01 = __builtin_amdgcn_wmma_f32_16x16x32_f16(false, a0, false, b1,
                                                       (short)0, acc01, false, false);
        acc10 = __builtin_amdgcn_wmma_f32_16x16x32_f16(false, a1, false, b0,
                                                       (short)0, acc10, false, false);
        acc11 = __builtin_amdgcn_wmma_f32_16x16x32_f16(false, a1, false, b1,
                                                       (short)0, acc11, false, false);
    }

    // ---- Epilogue: C/D layout M = vgpr + 8*half, N = lane%16 ----
    const int gn0 = n0 + nband + mrow;
    const int gn1 = n0 + nband + 16 + mrow;
    const float bb0 = bias ? bias[gn0] : 0.0f;
    const float bb1 = bias ? bias[gn1] : 0.0f;
#pragma unroll
    for (int r = 0; r < 8; ++r) {
        int gm0 = m0 + mband + r + 8 * half;
        int gm1 = gm0 + 16;
        float o;
        o = acc00[r] + bb0; C[(long)gm0 * ldc + gn0] = relu ? fmaxf(o, 0.0f) : o;
        o = acc01[r] + bb1; C[(long)gm0 * ldc + gn1] = relu ? fmaxf(o, 0.0f) : o;
        o = acc10[r] + bb0; C[(long)gm1 * ldc + gn0] = relu ? fmaxf(o, 0.0f) : o;
        o = acc11[r] + bb1; C[(long)gm1 * ldc + gn1] = relu ? fmaxf(o, 0.0f) : o;
    }
}

// ---------------------------------------------------------------------------
// Small helper kernels
// ---------------------------------------------------------------------------

// conv1_W [533,512] -> zero-padded [544,512]
__global__ void k_pad_conv1w(const float* __restrict__ W, float* __restrict__ Wp)
{
    int i = blockIdx.x * 256 + threadIdx.x;
    if (i >= 544 * 512) return;
    int r = i / 512, c = i % 512;
    Wp[i] = (r < 533) ? W[r * 512 + c] : 0.0f;
}

// h[:,0:21] = relu(x[:,:21] @ lin2_W + b) ; h[:,533:544] = 0
__global__ void k_lin2_pad(const float* __restrict__ x, const float* __restrict__ W2,
                           const float* __restrict__ b2, float* __restrict__ h)
{
    int row = blockIdx.x;       // 0..8191
    int t = threadIdx.x;        // 0..31
    const float* xr = x + (long)row * 6165;
    if (t < 21) {
        float s = b2[t];
        for (int j = 0; j < 21; ++j) s += xr[j] * W2[j * 21 + t];
        h[(long)row * 544 + t] = fmaxf(s, 0.0f);
    } else {
        h[(long)row * 544 + 512 + t] = 0.0f;   // cols 533..543
    }
}

// mode 0: dinv = rsqrt(max(rowsum+1, 1e-12))  (GCN with self loop)
// mode 1: dinv = 1/max(rowsum, 1)             (pool propagation)
__global__ void k_dinv(const float* __restrict__ adj, float* __restrict__ dinv,
                       int n, int mode)
{
    int b = blockIdx.y, i = blockIdx.x, t = threadIdx.x;
    const float* row = adj + ((long)b * n + i) * (long)n;
    float s = 0.0f;
    for (int j = t; j < n; j += 256) s += row[j];
    __shared__ float red[256];
    red[t] = s; __syncthreads();
    for (int o = 128; o > 0; o >>= 1) { if (t < o) red[t] += red[t + o]; __syncthreads(); }
    if (t == 0) {
        float d = red[0];
        dinv[(long)b * n + i] = (mode == 0) ? rsqrtf(fmaxf(d + 1.0f, 1e-12f))
                                            : 1.0f / fmaxf(d, 1.0f);
    }
}

// y[r,c] *= dinv[r]   (feature dim fixed 512)
__global__ void k_rowscale(float* __restrict__ y, const float* __restrict__ dinv, long rows)
{
    long i = (long)blockIdx.x * 256 + threadIdx.x;
    if (i >= rows * 512) return;
    y[i] *= dinv[i >> 9];
}

// out = relu(dinv*(t + y) + bias)   (A = adj + I folded as adj@y + y)
__global__ void k_gcn_epi(const float* __restrict__ t, const float* __restrict__ y,
                          const float* __restrict__ dinv, const float* __restrict__ bias,
                          float* __restrict__ out, long rows)
{
    long i = (long)blockIdx.x * 256 + threadIdx.x;
    if (i >= rows * 512) return;
    long r = i >> 9; int c = (int)(i & 511);
    out[i] = fmaxf(dinv[r] * (t[i] + y[i]) + bias[c], 0.0f);
}

// score[row] = sum_c | h[row,c] - dpool[row]*t[row,c] |
__global__ void k_score(const float* __restrict__ h, const float* __restrict__ t,
                        const float* __restrict__ dpool, float* __restrict__ score)
{
    long row = blockIdx.x; int tt = threadIdx.x;
    float dp = dpool[row];
    const float* hr = h + row * 512;
    const float* tr = t + row * 512;
    float s = 0.0f;
    for (int c = tt; c < 512; c += 256) s += fabsf(hr[c] - dp * tr[c]);
    __shared__ float red[256];
    red[tt] = s; __syncthreads();
    for (int o = 128; o > 0; o >>= 1) { if (tt < o) red[tt] += red[tt + o]; __syncthreads(); }
    if (tt == 0) score[row] = red[0];
}

// Per-graph bitonic sort (descending, tie-break lower index) -> keep top kk ids
__global__ void k_topk(const float* __restrict__ score, int* __restrict__ idx_out,
                       int n, int kk)
{
    __shared__ float sv[2048];
    __shared__ int   si[2048];
    int b = blockIdx.x, t = threadIdx.x, nt = blockDim.x;
    for (int i = t; i < n; i += nt) { sv[i] = score[(long)b * n + i]; si[i] = i; }
    __syncthreads();
    for (int ks = 2; ks <= n; ks <<= 1) {
        for (int js = ks >> 1; js > 0; js >>= 1) {
            for (int i = t; i < n; i += nt) {
                int l = i ^ js;
                if (l > i) {
                    float va = sv[i], vb = sv[l];
                    int ia = si[i], ib = si[l];
                    bool lPrec = (vb > va) || (vb == va && ib < ia);
                    bool up = ((i & ks) == 0);
                    if (up ? lPrec : !lPrec) {
                        sv[i] = vb; sv[l] = va; si[i] = ib; si[l] = ia;
                    }
                }
            }
            __syncthreads();
        }
    }
    for (int i = t; i < kk; i += nt) idx_out[b * kk + i] = si[i];
}

__global__ void k_gather_x(const float* __restrict__ h, const int* __restrict__ idx,
                           float* __restrict__ xk, int n, int kk)
{
    long i = (long)blockIdx.x * 256 + threadIdx.x;
    if (i >= (long)4 * kk * 512) return;
    int c = (int)(i & 511);
    long p = i >> 9;
    int b = (int)(p / kk), pp = (int)(p % kk);
    xk[i] = h[((long)b * n + idx[b * kk + pp]) * 512 + c];
}

__global__ void k_gather_adj(const float* __restrict__ adj, const int* __restrict__ idx,
                             float* __restrict__ adjk, int n, int kk)
{
    long i = (long)blockIdx.x * 256 + threadIdx.x;
    if (i >= (long)4 * kk * kk) return;
    int q = (int)(i % kk);
    long pq = i / kk;
    int p = (int)(pq % kk), b = (int)(pq / kk);
    adjk[i] = adj[((long)b * n + idx[b * kk + p]) * (long)n + idx[b * kk + q]];
}

// li = xk . att[:512], lj = xk . att[512:]
__global__ void k_lilj(const float* __restrict__ xk, const float* __restrict__ att,
                       float* __restrict__ li, float* __restrict__ lj, int kk)
{
    int row = blockIdx.x * 64 + threadIdx.x;
    if (row >= 4 * kk) return;
    const float* xr = xk + (long)row * 512;
    float a = 0.0f, c2 = 0.0f;
    for (int c = 0; c < 512; ++c) { a += xr[c] * att[c]; c2 += xr[c] * att[512 + c]; }
    li[row] = a; lj[row] = c2;
}

// new_adj[row,:] = softmax(li[row] + lj[b,:] + adjk[row,:])   (LAMB = 1.0)
__global__ void k_softmax_row(const float* __restrict__ li, const float* __restrict__ lj,
                              const float* __restrict__ adjk, float* __restrict__ outadj,
                              int kk)
{
    long row = blockIdx.x; int t = threadIdx.x;
    int b = (int)(row / kk);
    const float* ar = adjk + row * (long)kk;
    const float* ljb = lj + (long)b * kk;
    float lival = li[row];
    __shared__ float red[256];
    float mx = -1e30f;
    for (int q = t; q < kk; q += 256) mx = fmaxf(mx, lival + ljb[q] + ar[q]);
    red[t] = mx; __syncthreads();
    for (int o = 128; o > 0; o >>= 1) { if (t < o) red[t] = fmaxf(red[t], red[t + o]); __syncthreads(); }
    mx = red[0]; __syncthreads();
    float s = 0.0f;
    for (int q = t; q < kk; q += 256) s += __expf(lival + ljb[q] + ar[q] - mx);
    red[t] = s; __syncthreads();
    for (int o = 128; o > 0; o >>= 1) { if (t < o) red[t] += red[t + o]; __syncthreads(); }
    float inv = 1.0f / red[0];
    for (int q = t; q < kk; q += 256)
        outadj[row * (long)kk + q] = __expf(lival + ljb[q] + ar[q] - mx) * inv;
}

// r[b, f] = max_i x[b,i,f] ; r[b, 512+f] = mean_i x[b,i,f]
__global__ void k_readout(const float* __restrict__ x, float* __restrict__ r, int n)
{
    int f = blockIdx.x, b = blockIdx.y, t = threadIdx.x;
    const float* xb = x + (long)b * n * 512;
    float mx = -1e30f, s = 0.0f;
    for (int i = t; i < n; i += 256) {
        float v = xb[(long)i * 512 + f];
        mx = fmaxf(mx, v); s += v;
    }
    __shared__ float rm[256], rs[256];
    rm[t] = mx; rs[t] = s; __syncthreads();
    for (int o = 128; o > 0; o >>= 1) {
        if (t < o) { rm[t] = fmaxf(rm[t], rm[t + o]); rs[t] += rs[t + o]; }
        __syncthreads();
    }
    if (t == 0) { r[b * 1024 + f] = rm[0]; r[b * 1024 + 512 + f] = rs[0] / (float)n; }
}

__global__ void k_zsum(const float* __restrict__ r1, const float* __restrict__ r2,
                       const float* __restrict__ r3, float* __restrict__ z)
{
    int i = blockIdx.x * 256 + threadIdx.x;
    if (i >= 4096) return;
    z[i] = fmaxf(r1[i], 0.0f) + fmaxf(r2[i], 0.0f) + fmaxf(r3[i], 0.0f);
}

// Final MLP + log_softmax, one block (512 threads) per graph
__global__ void k_mlp(const float* __restrict__ z,
                      const float* __restrict__ W1, const float* __restrict__ b1,
                      const float* __restrict__ W2, const float* __restrict__ b2,
                      const float* __restrict__ W3, const float* __restrict__ b3,
                      float* __restrict__ out)
{
    __shared__ float zb[1024], s1[512], s2[256], ov[2];
    int b = blockIdx.x, t = threadIdx.x;
    for (int i = t; i < 1024; i += 512) zb[i] = z[b * 1024 + i];
    __syncthreads();
    {
        float s = b1[t];
        for (int j = 0; j < 1024; ++j) s += zb[j] * W1[j * 512 + t];
        s1[t] = fmaxf(s, 0.0f);
    }
    __syncthreads();
    if (t < 256) {
        float s = b2[t];
        for (int j = 0; j < 512; ++j) s += s1[j] * W2[j * 256 + t];
        s2[t] = fmaxf(s, 0.0f);
    }
    __syncthreads();
    if (t < 2) {
        float s = b3[t];
        for (int j = 0; j < 256; ++j) s += s2[j] * W3[j * 2 + t];
        ov[t] = s;
    }
    __syncthreads();
    if (t == 0) {
        float m = fmaxf(ov[0], ov[1]);
        float lse = m + logf(__expf(ov[0] - m) + __expf(ov[1] - m));
        out[b * 2 + 0] = ov[0] - lse;
        out[b * 2 + 1] = ov[1] - lse;
    }
}

// ---------------------------------------------------------------------------
// Host-side orchestration
// ---------------------------------------------------------------------------
extern "C" void kernel_launch(void* const* d_in, const int* in_sizes, int n_in,
                              void* d_out, int out_size, void* d_ws, size_t ws_size,
                              hipStream_t stream)
{
    (void)in_sizes; (void)n_in; (void)out_size; (void)ws_size;

    const float* x       = (const float*)d_in[0];
    const float* adj     = (const float*)d_in[1];
    const float* lin1_W  = (const float*)d_in[2];
    const float* lin1_b  = (const float*)d_in[3];
    const float* lin2_W  = (const float*)d_in[4];
    const float* lin2_b  = (const float*)d_in[5];
    const float* conv1_W = (const float*)d_in[6];
    const float* conv1_b = (const float*)d_in[7];
    const float* conv2_W = (const float*)d_in[8];
    const float* conv2_b = (const float*)d_in[9];
    const float* conv3_W = (const float*)d_in[10];
    const float* conv3_b = (const float*)d_in[11];
    const float* att1    = (const float*)d_in[12];
    const float* att2    = (const float*)d_in[13];
    const float* fc1_W   = (const float*)d_in[14];
    const float* fc1_b   = (const float*)d_in[15];
    const float* fc2_W   = (const float*)d_in[16];
    const float* fc2_b   = (const float*)d_in[17];
    const float* fc3_W   = (const float*)d_in[18];
    const float* fc3_b   = (const float*)d_in[19];
    float* outp = (float*)d_out;

    // Workspace layout (floats)
    const size_t OFF_HPAD  = 0;                               // 8192*544
    const size_t OFF_WPAD  = OFF_HPAD  + (size_t)8192 * 544;  // 544*512
    const size_t OFF_A     = OFF_WPAD  + (size_t)544 * 512;   // 8192*512 (xw / y)
    const size_t OFF_B     = OFF_A     + (size_t)8192 * 512;  // 8192*512 (adj@y / prop)
    const size_t OFF_C     = OFF_B     + (size_t)8192 * 512;  // 8192*512 (conv out)
    const size_t OFF_DINV  = OFF_C     + (size_t)8192 * 512;  // 8192
    const size_t OFF_DPOOL = OFF_DINV  + 8192;                // 8192
    const size_t OFF_SCORE = OFF_DPOOL + 8192;                // 8192
    const size_t OFF_IDX   = OFF_SCORE + 8192;                // 8192 (ints)
    const size_t OFF_XK1   = OFF_IDX   + 8192;                // 4*1024*512
    const size_t OFF_ADJK1 = OFF_XK1   + (size_t)4*1024*512;  // 4*1024*1024
    const size_t OFF_ADJ1  = OFF_ADJK1 + (size_t)4*1024*1024; // 4*1024*1024
    const size_t OFF_LI    = OFF_ADJ1  + (size_t)4*1024*1024; // 4096
    const size_t OFF_LJ    = OFF_LI    + 4096;
    const size_t OFF_R1    = OFF_LJ    + 4096;                // 4096 each
    const size_t OFF_R2    = OFF_R1    + 4096;
    const size_t OFF_R3    = OFF_R2    + 4096;
    const size_t OFF_XK2   = OFF_R3    + 4096;                // 4*512*512
    const size_t OFF_ADJK2 = OFF_XK2   + (size_t)4*512*512;
    const size_t OFF_ADJ2  = OFF_ADJK2 + (size_t)4*512*512;
    const size_t OFF_Z     = OFF_ADJ2  + (size_t)4*512*512;   // 4096

    float* ws    = (float*)d_ws;
    float* hpad  = ws + OFF_HPAD;
    float* wpad  = ws + OFF_WPAD;
    float* bufA  = ws + OFF_A;
    float* bufB  = ws + OFF_B;
    float* bufC  = ws + OFF_C;
    float* dinv  = ws + OFF_DINV;
    float* dpool = ws + OFF_DPOOL;
    float* score = ws + OFF_SCORE;
    int*   idx   = (int*)(ws + OFF_IDX);
    float* xk1   = ws + OFF_XK1;
    float* adjk1 = ws + OFF_ADJK1;
    float* adj1  = ws + OFF_ADJ1;
    float* li    = ws + OFF_LI;
    float* lj    = ws + OFF_LJ;
    float* r1    = ws + OFF_R1;
    float* r2    = ws + OFF_R2;
    float* r3    = ws + OFF_R3;
    float* xk2   = ws + OFF_XK2;
    float* adjk2 = ws + OFF_ADJK2;
    float* adj2  = ws + OFF_ADJ2;
    float* zbuf  = ws + OFF_Z;

    auto gemm = [&](const float* A, const float* Bm, float* C, int M, int N, int K,
                    long lda, long ldb, long ldc, long sA, long sB, long sC,
                    const float* bias, int relu, int batch) {
        dim3 g(N / 64, M / 128, batch);
        k_wmma_gemm<<<g, 256, 0, stream>>>(A, Bm, C, M, N, K, lda, ldb, ldc,
                                           sA, sB, sC, bias, relu);
    };

    // ---- Stage 0: feature projections into padded h [8192, 544] ----
    k_pad_conv1w<<<(544 * 512 + 255) / 256, 256, 0, stream>>>(conv1_W, wpad);
    k_lin2_pad<<<8192, 32, 0, stream>>>(x, lin2_W, lin2_b, hpad);
    // x1 = relu(x[:,21:] @ lin1_W + b) -> h cols 21..532
    gemm(x + 21, lin1_W, hpad + 21, 8192, 512, 6144, 6165, 512, 544, 0, 0, 0,
         lin1_b, 1, 1);

    // ---- conv1 (n = 2048) ----
    { dim3 g(2048, 4); k_dinv<<<g, 256, 0, stream>>>(adj, dinv, 2048, 0); }
    gemm(hpad, wpad, bufA, 8192, 512, 544, 544, 512, 512, 0, 0, 0, nullptr, 0, 1);
    k_rowscale<<<(8192 * 512 + 255) / 256, 256, 0, stream>>>(bufA, dinv, 8192);
    gemm(adj, bufA, bufB, 2048, 512, 2048, 2048, 512, 512,
         (long)2048 * 2048, (long)2048 * 512, (long)2048 * 512, nullptr, 0, 4);
    k_gcn_epi<<<(8192 * 512 + 255) / 256, 256, 0, stream>>>(bufB, bufA, dinv,
                                                            conv1_b, bufC, 8192);

    // ---- pool1 (2048 -> 1024) ----
    { dim3 g(2048, 4); k_dinv<<<g, 256, 0, stream>>>(adj, dpool, 2048, 1); }
    gemm(adj, bufC, bufB, 2048, 512, 2048, 2048, 512, 512,
         (long)2048 * 2048, (long)2048 * 512, (long)2048 * 512, nullptr, 0, 4);
    k_score<<<8192, 256, 0, stream>>>(bufC, bufB, dpool, score);
    k_topk<<<4, 1024, 0, stream>>>(score, idx, 2048, 1024);
    k_gather_x<<<(4 * 1024 * 512 + 255) / 256, 256, 0, stream>>>(bufC, idx, xk1, 2048, 1024);
    k_gather_adj<<<(4 * 1024 * 1024 + 255) / 256, 256, 0, stream>>>(adj, idx, adjk1, 2048, 1024);
    k_lilj<<<(4 * 1024 + 63) / 64, 64, 0, stream>>>(xk1, att1, li, lj, 1024);
    k_softmax_row<<<4 * 1024, 256, 0, stream>>>(li, lj, adjk1, adj1, 1024);
    { dim3 g(512, 4); k_readout<<<g, 256, 0, stream>>>(xk1, r1, 1024); }

    // ---- conv2 (n = 1024) ----
    { dim3 g(1024, 4); k_dinv<<<g, 256, 0, stream>>>(adj1, dinv, 1024, 0); }
    gemm(xk1, conv2_W, bufA, 4096, 512, 512, 512, 512, 512, 0, 0, 0, nullptr, 0, 1);
    k_rowscale<<<(4096 * 512 + 255) / 256, 256, 0, stream>>>(bufA, dinv, 4096);
    gemm(adj1, bufA, bufB, 1024, 512, 1024, 1024, 512, 512,
         (long)1024 * 1024, (long)1024 * 512, (long)1024 * 512, nullptr, 0, 4);
    k_gcn_epi<<<(4096 * 512 + 255) / 256, 256, 0, stream>>>(bufB, bufA, dinv,
                                                            conv2_b, bufC, 4096);

    // ---- pool2 (1024 -> 512) ----
    { dim3 g(1024, 4); k_dinv<<<g, 256, 0, stream>>>(adj1, dpool, 1024, 1); }
    gemm(adj1, bufC, bufB, 1024, 512, 1024, 1024, 512, 512,
         (long)1024 * 1024, (long)1024 * 512, (long)1024 * 512, nullptr, 0, 4);
    k_score<<<4096, 256, 0, stream>>>(bufC, bufB, dpool, score);
    k_topk<<<4, 512, 0, stream>>>(score, idx, 1024, 512);
    k_gather_x<<<(4 * 512 * 512 + 255) / 256, 256, 0, stream>>>(bufC, idx, xk2, 1024, 512);
    k_gather_adj<<<(4 * 512 * 512 + 255) / 256, 256, 0, stream>>>(adj1, idx, adjk2, 1024, 512);
    k_lilj<<<(4 * 512 + 63) / 64, 64, 0, stream>>>(xk2, att2, li, lj, 512);
    k_softmax_row<<<4 * 512, 256, 0, stream>>>(li, lj, adjk2, adj2, 512);
    { dim3 g(512, 4); k_readout<<<g, 256, 0, stream>>>(xk2, r2, 512); }

    // ---- conv3 (n = 512) ----
    { dim3 g(512, 4); k_dinv<<<g, 256, 0, stream>>>(adj2, dinv, 512, 0); }
    gemm(xk2, conv3_W, bufA, 2048, 512, 512, 512, 512, 512, 0, 0, 0, nullptr, 0, 1);
    k_rowscale<<<(2048 * 512 + 255) / 256, 256, 0, stream>>>(bufA, dinv, 2048);
    gemm(adj2, bufA, bufB, 512, 512, 512, 512, 512, 512,
         (long)512 * 512, (long)512 * 512, (long)512 * 512, nullptr, 0, 4);
    k_gcn_epi<<<(2048 * 512 + 255) / 256, 256, 0, stream>>>(bufB, bufA, dinv,
                                                            conv3_b, bufC, 2048);
    { dim3 g(512, 4); k_readout<<<g, 256, 0, stream>>>(bufC, r3, 512); }

    // ---- head ----
    k_zsum<<<(4096 + 255) / 256, 256, 0, stream>>>(r1, r2, r3, zbuf);
    k_mlp<<<4, 512, 0, stream>>>(zbuf, fc1_W, fc1_b, fc2_W, fc2_b, fc3_W, fc3_b, outp);
}
